// CarrierTokenAttention2DTimestep_7799660609901
// MI455X (gfx1250) — matrix operand order repaired and checked by reading, not compiled, not run on hardware
//
#include <hip/hip_runtime.h>

// ---------------------------------------------------------------------------
// CDNA5 (gfx1250) fused QKV-proj + 2D-RoPE + L2-norm + cosine attention.
// B=8, N=1024, DIM=1024, HEADS=16, HEAD_DIM=64.
// Matmuls via v_wmma_f32_16x16x32_f16 (wave32); tile staging via
// GLOBAL_LOAD_ASYNC_TO_LDS_B128 (ASYNCcnt-tracked) with a plain-copy fallback.
// ---------------------------------------------------------------------------

typedef __attribute__((ext_vector_type(16))) _Float16 v16h;
typedef __attribute__((ext_vector_type(8)))  float    v8f;
typedef int v4i __attribute__((vector_size(16)));     // matches builtin proto

union FragF { v16h h; float4 f[2]; };
union F4H8  { float4 f; _Float16 hv[8]; };

// ---- optional CDNA5 async global->LDS copy path ---------------------------
#if defined(__has_builtin)
#if __has_builtin(__builtin_amdgcn_global_load_async_to_lds_b128) && \
    __has_builtin(__builtin_amdgcn_s_wait_asynccnt)
#define HAVE_ASYNC_LDS 1
#endif
#endif

#ifdef HAVE_ASYNC_LDS
#define ASYNC_CP16(gsrc, ldst)                                              \
  __builtin_amdgcn_global_load_async_to_lds_b128(                           \
      (__attribute__((address_space(1))) v4i*)(gsrc),                       \
      (__attribute__((address_space(3))) v4i*)(ldst), 0, 0)
#define ASYNC_FENCE() __builtin_amdgcn_s_wait_asynccnt(0)
#else
#define ASYNC_CP16(gsrc, ldst) (*(ldst) = *(gsrc))
#define ASYNC_FENCE() ((void)0)
#endif

// A-operand fragment (16x32 f16, M x K) from an LDS tile stored row-major
// [rows][pitch]. ISA layout: lane half hh=lane/16, M=lane%16; VGPR 0-3 hold
// K = hh*8..hh*8+7 (contiguous), VGPR 4-7 hold K = 16+hh*8.. (contiguous).
// pitch*2 bytes must be a multiple of 16 (pitch = 72 here).
__device__ __forceinline__ v16h ldsFragA(const _Float16* lds, int row0, int pitch, int kOff) {
  const int lane = threadIdx.x & 31;
  const int m = lane & 15, hh = lane >> 4;
  const float4* p = (const float4*)(lds + (row0 + m) * pitch + kOff + hh * 8);
  FragF f;
  f.f[0] = p[0];    // K = kOff + hh*8      .. +7
  f.f[1] = p[2];    // K = kOff + 16 + hh*8 .. +7   (+16 halfs = +2 float4)
  return f.h;
}

// B-operand fragment (32x16 f16, K x N) from an "N-major" LDS tile:
// row n holds that output-column's K values (B[k][n] = tile[n][k]).
// ISA layout (same as A): VGPR 0-3 hold K = hh*8..hh*8+7, VGPR 4-7 hold
// K = 16+hh*8..16+hh*8+7.
__device__ __forceinline__ v16h ldsFragB(const _Float16* lds, int col0, int pitch, int kOff) {
  const int lane = threadIdx.x & 31;
  const int n = lane & 15, hh = lane >> 4;
  const float4* p = (const float4*)(lds + (col0 + n) * pitch + kOff + hh * 8);
  FragF f;
  f.f[0] = p[0];
  f.f[1] = p[2];
  return f.h;
}

// ---------------------------------------------------------------------------
// Kernel 1: fp32 -> f16 conversion (grid-stride)
// ---------------------------------------------------------------------------
__global__ void cvt_f32_f16(const float* __restrict__ src, _Float16* __restrict__ dst, int n) {
  int i = blockIdx.x * blockDim.x + threadIdx.x;
  const int stride = gridDim.x * blockDim.x;
  for (; i < n; i += stride) dst[i] = (_Float16)src[i];
}

// ---------------------------------------------------------------------------
// Kernel 2: qkv = x @ w^T  (M=8192, N=3072, K=1024), block tile 128x64,
// BK=64 (8 WMMAs per barrier pair). Fused epilogue: RoPE, L2-norm, and
// per-head logit scale folded into q. N-tile: 0..15 q / 16..31 k / 32..47 v.
// ---------------------------------------------------------------------------
__global__ __launch_bounds__(256) void qkv_gemm_rope(
    const _Float16* __restrict__ xh, const _Float16* __restrict__ wh,
    const float* __restrict__ lsg,
    _Float16* __restrict__ qh, _Float16* __restrict__ kh, _Float16* __restrict__ vh)
{
  constexpr int TP = 72;   // LDS pitch in halfs (144B rows, 16B aligned)
  constexpr int AP = 65;   // fp32 accum pitch (floats)
  __shared__ float smemF[128 * AP];           // 33.3 KB, aliased across phases
  _Float16* As = (_Float16*)smemF;            // [128][72] f16 (64 K used)
  _Float16* Bs = As + 128 * TP;               // [64][72]

  const int ntile = blockIdx.x;               // 0..47
  const int mtile = blockIdx.y;               // 0..63
  const int tid   = threadIdx.x;
  const int wid   = tid >> 5;
  const int lane  = tid & 31;
  const int hh    = lane >> 4, nn = lane & 15;
  const int wm    = wid & 3;                  // 4 waves along M (32 rows each)
  const int wn    = wid >> 2;                 // 2 waves along N (32 cols each)

  v8f zero = {};
  v8f acc[2][2];
  acc[0][0] = zero; acc[0][1] = zero; acc[1][0] = zero; acc[1][1] = zero;

  const float4* xg = (const float4*)xh;       // 8 halfs per float4
  const float4* wg = (const float4*)wh;
  float4* AsV = (float4*)As;                  // pitch 72 halfs = 9 float4
  float4* BsV = (float4*)Bs;

  for (int k0 = 0; k0 < 1024; k0 += 64) {
#pragma unroll
    for (int i = 0; i < 4; ++i) {             // A: 128x64 halfs = 1024 float4
      const int idx = tid + 256 * i;
      const int row = idx >> 3, ks = idx & 7;
      ASYNC_CP16(&xg[(size_t)(mtile * 128 + row) * 128 + (k0 >> 3) + ks],
                 &AsV[row * 9 + ks]);
    }
#pragma unroll
    for (int i = 0; i < 2; ++i) {             // B: 64x64 halfs = 512 float4
      const int idx = tid + 256 * i;
      const int row = idx >> 3, ks = idx & 7;
      ASYNC_CP16(&wg[(size_t)(ntile * 64 + row) * 128 + (k0 >> 3) + ks],
                 &BsV[row * 9 + ks]);
    }
    ASYNC_FENCE();
    __syncthreads();

#pragma unroll
    for (int kk = 0; kk < 64; kk += 32) {
      v16h a0 = ldsFragA(As, wm * 32,      TP, kk);
      v16h a1 = ldsFragA(As, wm * 32 + 16, TP, kk);
      v16h b0 = ldsFragB(Bs, wn * 32,      TP, kk);
      v16h b1 = ldsFragB(Bs, wn * 32 + 16, TP, kk);
      acc[0][0] = __builtin_amdgcn_wmma_f32_16x16x32_f16(false, a0, false, b0, (short)0, acc[0][0], false, false);
      acc[0][1] = __builtin_amdgcn_wmma_f32_16x16x32_f16(false, a0, false, b1, (short)0, acc[0][1], false, false);
      acc[1][0] = __builtin_amdgcn_wmma_f32_16x16x32_f16(false, a1, false, b0, (short)0, acc[1][0], false, false);
      acc[1][1] = __builtin_amdgcn_wmma_f32_16x16x32_f16(false, a1, false, b1, (short)0, acc[1][1], false, false);
    }
    __syncthreads();
  }

  // Spill fp32 accumulators to LDS (C layout: VGPR r -> M = r + 8*(lane/16))
  float* accF = smemF;
#pragma unroll
  for (int i = 0; i < 2; ++i)
#pragma unroll
    for (int j = 0; j < 2; ++j)
#pragma unroll
      for (int r = 0; r < 8; ++r)
        accF[(wm * 32 + i * 16 + hh * 8 + r) * AP + wn * 32 + j * 16 + nn] = acc[i][j][r];
  __syncthreads();

  // Per-row epilogue: one thread per (token, head) row of 64 values.
  if (tid < 128) {
    const int row = tid;
    const int mg  = mtile * 128 + row;
    const int b   = mg >> 10, tok = mg & 1023;
    const int kind = ntile >> 4;              // 0=q, 1=k, 2=v
    const int head = ntile & 15;
    float vals[64];
#pragma unroll
    for (int d = 0; d < 64; ++d) vals[d] = accF[row * AP + d];

    if (kind < 2) {
      // 2D axial RoPE: j<16 uses grid-row tok/32, j>=16 uses grid-col tok%32,
      // freq_i = 10000^(-4i/64). Then L2-normalize; fold exp(min(ls,clamp))
      // into q.
      const float LOG1E4 = 9.210340371976184f;
      const int ph = tok >> 5, pw = tok & 31;
      for (int j = 0; j < 32; ++j) {
        const int   fi  = (j < 16) ? j : (j - 16);
        const float pos = (j < 16) ? (float)ph : (float)pw;
        const float ang = pos * __expf((float)fi * (-4.0f / 64.0f) * LOG1E4);
        float s, c;
        __sincosf(ang, &s, &c);
        const float xr = vals[2 * j], xi = vals[2 * j + 1];
        vals[2 * j]     = xr * c - xi * s;
        vals[2 * j + 1] = xr * s + xi * c;
      }
      float ss = 0.f;
#pragma unroll
      for (int d = 0; d < 64; ++d) ss += vals[d] * vals[d];
      float inv = 1.0f / fmaxf(sqrtf(ss), 1e-12f);
      if (kind == 0) inv *= __expf(fminf(lsg[head], 4.6052f));  // logit scale
#pragma unroll
      for (int d = 0; d < 64; ++d) vals[d] *= inv;
    }
    _Float16* dst = (kind == 0 ? qh : (kind == 1 ? kh : vh)) +
                    ((((size_t)b * 16 + head) * 1024 + tok) * 64);
#pragma unroll
    for (int d = 0; d < 64; ++d) dst[d] = (_Float16)vals[d];
  }
}

// ---------------------------------------------------------------------------
// Kernel 3: flash attention. Block = (q-tile 64, head, batch), 8 waves.
// Online softmax state (m,l) lives in registers of lanes 0..63.
// ---------------------------------------------------------------------------
__global__ __launch_bounds__(256) void attn_kernel(
    const _Float16* __restrict__ qh, const _Float16* __restrict__ kh,
    const _Float16* __restrict__ vh, float* __restrict__ out)
{
  constexpr int TP = 72;                      // f16 tile pitch: 144B
  __shared__ _Float16 Qs[64 * TP];
  __shared__ _Float16 Ks[64 * TP];
  __shared__ _Float16 Vt[64 * TP];            // V transposed: [d][kv]
  __shared__ _Float16 Pl[64 * TP];            // P (exp) as f16: [q][kv]
  __shared__ float    Sl[64 * 65];            // logits fp32 (scale pre-folded)
  __shared__ float    fac[64];                // per-row rescale factor
  __shared__ float    lsh[64];                // final row sums

  const int qt  = blockIdx.x, h = blockIdx.y, b = blockIdx.z;
  const int tid = threadIdx.x;
  const int wid = tid >> 5, lane = tid & 31;
  const int hh  = lane >> 4, nn = lane & 15;
  const int tr  = wid & 3;                    // wave's tile row (16 q rows)
  const int tcp = wid >> 2;                   // wave's tile-column pair
  const int m0  = tr * 16;

  const size_t headBase = (((size_t)b * 16 + h) * 1024) * 64;

  // Stage resident Q tile
  const float4* qg = (const float4*)(qh + headBase);
  float4* QsV = (float4*)Qs;
#pragma unroll
  for (int i = 0; i < 2; ++i) {
    const int idx = tid + 256 * i;
    const int row = idx >> 3, ks = idx & 7;
    ASYNC_CP16(&qg[(size_t)(qt * 64 + row) * 8 + ks], &QsV[row * 9 + ks]);
  }
  ASYNC_FENCE();
  __syncthreads();

  // Q fragments are loop-invariant: hoist out of the KV loop.
  const v16h qa0 = ldsFragA(Qs, m0, TP, 0);
  const v16h qa1 = ldsFragA(Qs, m0, TP, 32);

  float mrow = -1e30f, lrow = 0.f;            // used by lanes with tid<64
  v8f zero = {};
  v8f oacc[2];
  oacc[0] = zero; oacc[1] = zero;

  const float4* kg = (const float4*)(kh + headBase);
  const float4* vg = (const float4*)(vh + headBase);
  float4* KsV = (float4*)Ks;

  for (int t = 0; t < 16; ++t) {
    __syncthreads();                          // prior readers of Ks/Vt done
#pragma unroll
    for (int i = 0; i < 2; ++i) {
      const int idx = tid + 256 * i;
      const int row = idx >> 3, ks = idx & 7; // row = kv index in tile
      ASYNC_CP16(&kg[(size_t)(t * 64 + row) * 8 + ks], &KsV[row * 9 + ks]);
      F4H8 u; u.f = vg[(size_t)(t * 64 + row) * 8 + ks];   // manual transpose
#pragma unroll
      for (int e = 0; e < 8; ++e) Vt[(ks * 8 + e) * TP + row] = u.hv[e];
    }
    ASYNC_FENCE();
    __syncthreads();

    // S = Q Kt^T   (64x64; each wave computes 2 tiles of 16x16)
#pragma unroll
    for (int jj = 0; jj < 2; ++jj) {
      const int c0 = (tcp * 2 + jj) * 16;
      v16h b0 = ldsFragB(Ks, c0, TP, 0);
      v16h b1 = ldsFragB(Ks, c0, TP, 32);
      v8f s = zero;
      s = __builtin_amdgcn_wmma_f32_16x16x32_f16(false, qa0, false, b0, (short)0, s, false, false);
      s = __builtin_amdgcn_wmma_f32_16x16x32_f16(false, qa1, false, b1, (short)0, s, false, false);
#pragma unroll
      for (int r = 0; r < 8; ++r)
        Sl[(m0 + hh * 8 + r) * 65 + c0 + nn] = s[r];
    }
    __syncthreads();

    // Online softmax row update; emit P as f16
    if (tid < 64) {
      const int row = tid;
      float mx = mrow;
      for (int j = 0; j < 64; ++j) mx = fmaxf(mx, Sl[row * 65 + j]);
      float sum = 0.f;
      for (int j = 0; j < 64; ++j) {
        const float p = __expf(Sl[row * 65 + j] - mx);
        sum += p;
        Pl[row * TP + j] = (_Float16)p;
      }
      const float f = __expf(mrow - mx);
      lrow = lrow * f + sum;
      mrow = mx;
      fac[row] = f;
    }
    __syncthreads();

    // O = O*f + P V   (each wave: same 2 tiles, cols = head-dim)
    {
      v16h a0 = ldsFragA(Pl, m0, TP, 0);
      v16h a1 = ldsFragA(Pl, m0, TP, 32);
#pragma unroll
      for (int jj = 0; jj < 2; ++jj) {
        const int c0 = (tcp * 2 + jj) * 16;
#pragma unroll
        for (int r = 0; r < 8; ++r)
          oacc[jj][r] *= fac[m0 + hh * 8 + r];
        v16h b0 = ldsFragB(Vt, c0, TP, 0);
        v16h b1 = ldsFragB(Vt, c0, TP, 32);
        oacc[jj] = __builtin_amdgcn_wmma_f32_16x16x32_f16(false, a0, false, b0, (short)0, oacc[jj], false, false);
        oacc[jj] = __builtin_amdgcn_wmma_f32_16x16x32_f16(false, a1, false, b1, (short)0, oacc[jj], false, false);
      }
    }
  }

  __syncthreads();
  if (tid < 64) lsh[tid] = lrow;
  __syncthreads();

  // out[b, tok, h*64 + d] = O / l
#pragma unroll
  for (int jj = 0; jj < 2; ++jj) {
    const int c0 = (tcp * 2 + jj) * 16;
#pragma unroll
    for (int r = 0; r < 8; ++r) {
      const int rowL = m0 + hh * 8 + r;
      const int tok  = qt * 64 + rowL;
      out[(((size_t)b * 1024) + tok) * 1024 + h * 64 + c0 + nn] = oacc[jj][r] / lsh[rowL];
    }
  }
}

// ---------------------------------------------------------------------------
extern "C" void kernel_launch(void* const* d_in, const int* in_sizes, int n_in,
                              void* d_out, int out_size, void* d_ws, size_t ws_size,
                              hipStream_t stream) {
  (void)in_sizes; (void)n_in; (void)out_size; (void)ws_size;
  const float* x  = (const float*)d_in[0];   // (8,1024,1024)
  const float* w  = (const float*)d_in[1];   // (3072,1024)
  const float* ls = (const float*)d_in[2];   // (16,1,1)
  float* out = (float*)d_out;                // (8,1024,1024)

  char* ws = (char*)d_ws;
  size_t off = 0;
  _Float16* xh = (_Float16*)(ws + off); off += (size_t)8 * 1024 * 1024 * 2;   // 16 MB
  _Float16* wh = (_Float16*)(ws + off); off += (size_t)3072 * 1024 * 2;       //  6 MB
  _Float16* qh = (_Float16*)(ws + off); off += (size_t)8 * 16 * 1024 * 64 * 2;
  _Float16* kh = (_Float16*)(ws + off); off += (size_t)8 * 16 * 1024 * 64 * 2;
  _Float16* vh = (_Float16*)(ws + off); off += (size_t)8 * 16 * 1024 * 64 * 2;

  cvt_f32_f16<<<2048, 256, 0, stream>>>(x, xh, 8 * 1024 * 1024);
  cvt_f32_f16<<<1024, 256, 0, stream>>>(w, wh, 3072 * 1024);
  qkv_gemm_rope<<<dim3(48, 64), 256, 0, stream>>>(xh, wh, ls, qh, kh, vh);
  attn_kernel<<<dim3(16, 16, 8), 256, 0, stream>>>(qh, kh, vh, out);
}
